// SecureResidualFeedForward_70214125355101
// MI455X (gfx1250) — matrix operand order repaired and checked by reading, not compiled, not run on hardware
//
#include <hip/hip_runtime.h>

typedef __attribute__((ext_vector_type(16))) __bf16 v16bf;
typedef __attribute__((ext_vector_type(8)))  __bf16 v8bf;
typedef __attribute__((ext_vector_type(8)))  float  v8f;

#define TILE_M 128
#define TILE_N 128
#define TILE_K 32
#define BKP    40   // LDS pitch (elements): 32 + 8 pad to break bank conflicts

// ---------------------------------------------------------------------------
// CDNA5 async global->LDS copy path (guarded; host pass / older toolchains
// fall back to register-staged copies).
// ROCm 7.2 signature (from diagnostics): params are pointers to int4 vectors.
// ---------------------------------------------------------------------------
#if defined(__AMDGCN__) && __has_builtin(__builtin_amdgcn_global_load_async_to_lds_b128)
#define USE_ASYNC_LDS 1
typedef int v4i_async __attribute__((vector_size(4 * sizeof(int))));
__device__ __forceinline__ void async_copy_b128(const void* g, void* l) {
    __builtin_amdgcn_global_load_async_to_lds_b128(
        (v4i_async*)g, (v4i_async*)l, 0, 0);
}
__device__ __forceinline__ void async_wait0() {
#if __has_builtin(__builtin_amdgcn_s_wait_asynccnt)
    __builtin_amdgcn_s_wait_asynccnt(0);
#else
    asm volatile("s_wait_asynccnt 0x0" ::: "memory");
#endif
}
#else
#define USE_ASYNC_LDS 0
#endif

__device__ __forceinline__ float gelu_poly(float x) {
    float z  = 1.702f * x;
    float z2 = z * z;
    float s  = 0.5f + 0.25f * z - 0.0208f * (z2 * z);
    return x * s;
}

// ---------------------------------------------------------------------------
// fp32 -> bf16 conversion (vectorized by 4)
// ---------------------------------------------------------------------------
__global__ __launch_bounds__(256) void f32_to_bf16_kernel(
    const float* __restrict__ in, __bf16* __restrict__ out, int n) {
    int i = (blockIdx.x * blockDim.x + threadIdx.x) * 4;
    if (i + 3 < n) {
        float4 v = *(const float4*)(in + i);
        out[i + 0] = (__bf16)v.x;
        out[i + 1] = (__bf16)v.y;
        out[i + 2] = (__bf16)v.z;
        out[i + 3] = (__bf16)v.w;
    } else {
        for (int k = i; k < n; ++k) out[k] = (__bf16)in[k];
    }
}

// ---------------------------------------------------------------------------
// fp32 (K x N) -> bf16 transposed (N x K). LDS-tiled 32x32 for coalescing.
// Block: 256 threads. Grid: (N/32, K/32).
// ---------------------------------------------------------------------------
__global__ __launch_bounds__(256) void f32_to_bf16_transpose_kernel(
    const float* __restrict__ in, __bf16* __restrict__ out, int K, int N) {
    __shared__ float tile[32][33];
    int k0 = blockIdx.y * 32, n0 = blockIdx.x * 32;
    int tx = threadIdx.x & 31;
    int ty = threadIdx.x >> 5;  // 0..7
#pragma unroll
    for (int j = 0; j < 32; j += 8)
        tile[ty + j][tx] = in[(size_t)(k0 + ty + j) * N + n0 + tx];
    __syncthreads();
#pragma unroll
    for (int j = 0; j < 32; j += 8)
        out[(size_t)(n0 + ty + j) * K + k0 + tx] = (__bf16)tile[tx][ty + j];
}

// ---------------------------------------------------------------------------
// bf16 WMMA GEMM: C = A(MxK) * Bt^T + bias, Bt is N x K row-major
// (pre-transposed weights). Double-buffered LDS, async global->LDS staging.
// Block: 256 threads = 8 waves. Tile 128x128x32. Wave = 32x64 patch.
// ---------------------------------------------------------------------------
template <bool GELU, bool RESID>
__global__ __launch_bounds__(256) void gemm_bf16_wmma(
    const __bf16* __restrict__ A,      // M x K row-major, bf16
    const __bf16* __restrict__ Bt,     // N x K row-major, bf16 (B transposed)
    const float*  __restrict__ bias,   // N, fp32
    const float*  __restrict__ resid,  // M x N fp32 (RESID only)
    __bf16* __restrict__ Cbf,          // output if !RESID
    float*  __restrict__ Cf,           // output if RESID
    int M, int N, int K) {

    __shared__ __bf16 As[2][TILE_M * BKP];   // [m][k]
    __shared__ __bf16 Bs[2][TILE_N * BKP];   // [n][k]

    const int t    = threadIdx.x;
    const int lane = t & 31;
    const int wid  = t >> 5;
    const int wm   = wid & 3;   // 4 waves along M
    const int wn   = wid >> 2;  // 2 waves along N

    const int rowBase = blockIdx.y * TILE_M;
    const int colBase = blockIdx.x * TILE_N;

    // Copy assignment: each thread moves 2 x b128 per tile per matrix.
    const int cRow  = t >> 2;          // 0..63 (two passes of 64 rows)
    const int cCol8 = (t & 3) * 8;     // 0,8,16,24

    const __bf16* aBase = A  + (size_t)(rowBase + cRow) * K + cCol8;
    const __bf16* bBase = Bt + (size_t)(colBase + cRow) * K + cCol8;

    v8f acc[2][4];
#pragma unroll
    for (int mi = 0; mi < 2; ++mi)
#pragma unroll
        for (int ni = 0; ni < 4; ++ni)
            acc[mi][ni] = (v8f){0.f, 0.f, 0.f, 0.f, 0.f, 0.f, 0.f, 0.f};

    const int ktiles = K / TILE_K;

#if USE_ASYNC_LDS
    auto stage = [&](int buf, int kbase) {
#pragma unroll
        for (int p = 0; p < 2; ++p) {
            async_copy_b128(aBase + (size_t)p * 64 * K + kbase,
                            &As[buf][(p * 64 + cRow) * BKP + cCol8]);
            async_copy_b128(bBase + (size_t)p * 64 * K + kbase,
                            &Bs[buf][(p * 64 + cRow) * BKP + cCol8]);
        }
    };
#else
    v8bf ra[2], rb[2];
    auto stage_load = [&](int kbase) {
#pragma unroll
        for (int p = 0; p < 2; ++p) {
            ra[p] = *(const v8bf*)(aBase + (size_t)p * 64 * K + kbase);
            rb[p] = *(const v8bf*)(bBase + (size_t)p * 64 * K + kbase);
        }
    };
    auto stage_store = [&](int buf) {
#pragma unroll
        for (int p = 0; p < 2; ++p) {
            *(v8bf*)&As[buf][(p * 64 + cRow) * BKP + cCol8] = ra[p];
            *(v8bf*)&Bs[buf][(p * 64 + cRow) * BKP + cCol8] = rb[p];
        }
    };
#endif

    // 8x v_wmma_f32_16x16x32_bf16 on one K-tile from LDS buffer `buf`.
    auto compute = [&](int buf) {
        const int khalf = lane >> 4;
        const int l16   = lane & 15;
        v16bf afrag[2];
#pragma unroll
        for (int mi = 0; mi < 2; ++mi) {
            int mrow = wm * 32 + mi * 16 + l16;
            union { v16bf v; v8bf h[2]; } u;
            // A frag (16-bit, 16x32): VGPR0-3 K=khalf*8..+7, VGPR4-7 K=16+khalf*8..+7
            u.h[0] = *(const v8bf*)&As[buf][mrow * BKP + khalf * 8];
            u.h[1] = *(const v8bf*)&As[buf][mrow * BKP + 16 + khalf * 8];
            afrag[mi] = u.v;
        }
        v16bf bfrag[4];
#pragma unroll
        for (int ni = 0; ni < 4; ++ni) {
            int ncol = wn * 64 + ni * 16 + l16;
            union { v16bf v; v8bf h[2]; } u;
            // B frag (32x16): VGPR0-3 K=khalf*8..+7, VGPR4-7 K=16+khalf*8..+7
            u.h[0] = *(const v8bf*)&Bs[buf][ncol * BKP + khalf * 8];
            u.h[1] = *(const v8bf*)&Bs[buf][ncol * BKP + 16 + khalf * 8];
            bfrag[ni] = u.v;
        }
#pragma unroll
        for (int mi = 0; mi < 2; ++mi)
#pragma unroll
            for (int ni = 0; ni < 4; ++ni)
                acc[mi][ni] = __builtin_amdgcn_wmma_f32_16x16x32_bf16(
                    false, afrag[mi], false, bfrag[ni],
                    (short)0, acc[mi][ni], false, false);
    };

#if USE_ASYNC_LDS
    // Pipeline: wait(own async) -> barrier -> issue async for kt+1 -> WMMA kt.
    stage(0, 0);
    int cur = 0;
    for (int kt = 0; kt < ktiles; ++kt) {
        async_wait0();
        __syncthreads();
        if (kt + 1 < ktiles) stage(cur ^ 1, (kt + 1) * TILE_K);
        compute(cur);
        cur ^= 1;
    }
#else
    // Register-staged double buffer: loads for kt+1 issued before the barrier,
    // LDS stores after the WMMAs so load latency hides behind compute.
    stage_load(0);
    stage_store(0);
    int cur = 0;
    for (int kt = 0; kt < ktiles; ++kt) {
        if (kt + 1 < ktiles) stage_load((kt + 1) * TILE_K);
        __syncthreads();
        compute(cur);
        if (kt + 1 < ktiles) stage_store(cur ^ 1);
        cur ^= 1;
    }
#endif

    // ---- epilogue: bias (+GELU) (+residual) ----
    const int l16   = lane & 15;
    const int rhalf = (lane >> 4) * 8;  // C/D layout: row = j + 8*(lane>=16)
#pragma unroll
    for (int mi = 0; mi < 2; ++mi) {
#pragma unroll
        for (int ni = 0; ni < 4; ++ni) {
            int r0 = rowBase + wm * 32 + mi * 16 + rhalf;
            int c  = colBase + wn * 64 + ni * 16 + l16;
            float bz = bias[c];
#pragma unroll
            for (int j = 0; j < 8; ++j) {
                float v = acc[mi][ni][j] + bz;
                if (GELU) v = gelu_poly(v);
                size_t idx = (size_t)(r0 + j) * N + c;
                if (RESID) {
                    Cf[idx] = v + resid[idx];
                } else {
                    Cbf[idx] = (__bf16)v;
                }
            }
        }
    }
}

// ---------------------------------------------------------------------------
// In-place LayerNorm over rows of H=1024 fp32. One block (256 thr) per row.
// ---------------------------------------------------------------------------
__device__ __forceinline__ float block_sum256(float v, float* red) {
#pragma unroll
    for (int o = 16; o > 0; o >>= 1) v += __shfl_down(v, o, 32);
    int lane = threadIdx.x & 31, wid = threadIdx.x >> 5;
    if (lane == 0) red[wid] = v;
    __syncthreads();
    float r;
    if (wid == 0) {
        r = (lane < 8) ? red[lane] : 0.f;
#pragma unroll
        for (int o = 4; o > 0; o >>= 1) r += __shfl_down(r, o, 32);
        if (lane == 0) red[0] = r;
    }
    __syncthreads();
    r = red[0];
    __syncthreads();
    return r;
}

__global__ __launch_bounds__(256) void layernorm_kernel(
    float* __restrict__ io, const float* __restrict__ w,
    const float* __restrict__ b, int H) {
    __shared__ float red[8];
    float* p = io + (size_t)blockIdx.x * H;
    int c = threadIdx.x * 4;

    float4 v = *(const float4*)(p + c);
    float s = v.x + v.y + v.z + v.w;
    float mu = block_sum256(s, red) * (1.0f / 1024.0f);

    float dx = v.x - mu, dy = v.y - mu, dz = v.z - mu, dw = v.w - mu;
    float s2 = dx * dx + dy * dy + dz * dz + dw * dw;
    float var = block_sum256(s2, red) * (1.0f / 1024.0f);
    float rstd = rsqrtf(var + 1e-12f);

    float4 wv = *(const float4*)(w + c);
    float4 bv = *(const float4*)(b + c);
    float4 o;
    o.x = dx * rstd * wv.x + bv.x;
    o.y = dy * rstd * wv.y + bv.y;
    o.z = dz * rstd * wv.z + bv.z;
    o.w = dw * rstd * wv.w + bv.w;
    *(float4*)(p + c) = o;
}

// ---------------------------------------------------------------------------
extern "C" void kernel_launch(void* const* d_in, const int* in_sizes, int n_in,
                              void* d_out, int out_size, void* d_ws, size_t ws_size,
                              hipStream_t stream) {
    (void)in_sizes; (void)n_in; (void)out_size; (void)ws_size;

    const float* x   = (const float*)d_in[0];
    const float* W1  = (const float*)d_in[1];
    const float* b1  = (const float*)d_in[2];
    const float* W2  = (const float*)d_in[3];
    const float* b2  = (const float*)d_in[4];
    const float* W3  = (const float*)d_in[5];
    const float* b3  = (const float*)d_in[6];
    const float* lnw = (const float*)d_in[7];
    const float* lnb = (const float*)d_in[8];

    const int M = 4 * 4096;   // B*S tokens
    const int H = 1024;
    const int I = 4096;

    char*  ws  = (char*)d_ws;
    size_t off = 0;
    auto carve = [&](size_t bytes) -> char* {
        char* p = ws + off;
        off += (bytes + 255) & ~(size_t)255;
        return p;
    };
    __bf16* xbf  = (__bf16*)carve((size_t)M * H * 2);
    __bf16* w1t  = (__bf16*)carve((size_t)H * I * 2);   // I x H
    __bf16* w2t  = (__bf16*)carve((size_t)I * I * 2);   // I x I
    __bf16* w3t  = (__bf16*)carve((size_t)I * H * 2);   // H x I
    __bf16* h1bf = (__bf16*)carve((size_t)M * I * 2);
    __bf16* h2bf = (__bf16*)carve((size_t)M * I * 2);

    // x -> bf16 (row-major, A-side)
    {
        size_t n = (size_t)M * H;
        f32_to_bf16_kernel<<<(int)((n / 4 + 255) / 256), 256, 0, stream>>>(x, xbf, (int)n);
    }
    // Weights -> bf16 transposed (N x K) so B tiles are contiguous rows.
    f32_to_bf16_transpose_kernel<<<dim3(I / 32, H / 32), 256, 0, stream>>>(W1, w1t, H, I);
    f32_to_bf16_transpose_kernel<<<dim3(I / 32, I / 32), 256, 0, stream>>>(W2, w2t, I, I);
    f32_to_bf16_transpose_kernel<<<dim3(H / 32, I / 32), 256, 0, stream>>>(W3, w3t, I, H);

    dim3 blk(256);
    // GEMM1: h1 = gelu(x @ W1 + b1)
    gemm_bf16_wmma<true, false><<<dim3(I / TILE_N, M / TILE_M), blk, 0, stream>>>(
        xbf, w1t, b1, nullptr, h1bf, nullptr, M, I, H);
    // GEMM2: h2 = gelu(h1 @ W2 + b2)
    gemm_bf16_wmma<true, false><<<dim3(I / TILE_N, M / TILE_M), blk, 0, stream>>>(
        h1bf, w2t, b2, nullptr, h2bf, nullptr, M, I, I);
    // GEMM3: pre = h2 @ W3 + b3 + x   (fp32 out to d_out)
    gemm_bf16_wmma<false, true><<<dim3(H / TILE_N, M / TILE_M), blk, 0, stream>>>(
        h2bf, w3t, b3, x, nullptr, (float*)d_out, M, H, I);
    // LayerNorm in-place on d_out
    layernorm_kernel<<<M, 256, 0, stream>>>((float*)d_out, lnw, lnb, H);
}
